// MaskedSelfAttention_24240795418716
// MI455X (gfx1250) — matrix operand order repaired and checked by reading, not compiled, not run on hardware
//
#include <hip/hip_runtime.h>

#define SEQ 4096
#define DM  2048

typedef __bf16 bf16;
typedef __attribute__((ext_vector_type(16))) __bf16 v16bf;
typedef __attribute__((ext_vector_type(8)))  __bf16 v8bf;
typedef __attribute__((ext_vector_type(8)))  float  v8f;

// ---------------------------------------------------------------- helpers

static __device__ __forceinline__ v16bf cat16(v8bf x, v8bf y) {
  return __builtin_shufflevector(x, y, 0,1,2,3,4,5,6,7,8,9,10,11,12,13,14,15);
}

static __device__ __forceinline__ v16bf pack16(float4 f0, float4 f1,
                                               float4 f2, float4 f3) {
  v16bf v;
  v[0]=(bf16)f0.x;  v[1]=(bf16)f0.y;  v[2]=(bf16)f0.z;  v[3]=(bf16)f0.w;
  v[4]=(bf16)f1.x;  v[5]=(bf16)f1.y;  v[6]=(bf16)f1.z;  v[7]=(bf16)f1.w;
  v[8]=(bf16)f2.x;  v[9]=(bf16)f2.y;  v[10]=(bf16)f2.z; v[11]=(bf16)f2.w;
  v[12]=(bf16)f3.x; v[13]=(bf16)f3.y; v[14]=(bf16)f3.z; v[15]=(bf16)f3.w;
  return v;
}

// A-matrix 16x32 bf16 fragment: lane holds row m=lane&15.
// K map: contiguous K run: k = k0 + 2*hi8 + h
static __device__ __forceinline__ v16bf load_a_f32(const float* p, int hi8) {
  const float4* q0 = (const float4*)(p + 2 * hi8);
  const float4* q1 = (const float4*)(p + 2 * hi8 + 8);
  return pack16(q0[0], q0[1], q1[0], q1[1]);
}
static __device__ __forceinline__ v16bf load_a_bf16(const bf16* p, int hi8) {
  v8bf x = *(const v8bf*)(p + 2 * hi8);
  v8bf y = *(const v8bf*)(p + 2 * hi8 + 8);
  return cat16(x, y);
}
// B-matrix 32x16 bf16 fragment: lane holds column n=lane&15,
// contiguous K run: k = k0 + hi16 + h  (hi16 = 16 for lanes 16..31)
static __device__ __forceinline__ v16bf load_b_f32(const float* p, int hi16) {
  const float4* q = (const float4*)(p + hi16);
  return pack16(q[0], q[1], q[2], q[3]);
}
static __device__ __forceinline__ v16bf load_b_bf16(const bf16* p, int hi16) {
  v8bf x = *(const v8bf*)(p + hi16);
  v8bf y = *(const v8bf*)(p + hi16 + 8);
  return cat16(x, y);
}

static __device__ __forceinline__ v8f wmma_bf16(v16bf a, v16bf b, v8f c) {
  return __builtin_amdgcn_wmma_f32_16x16x32_bf16(
      false, a, false, b, (short)0, c, false, false);
}

// ---------------------------------------------------------------- phase 1
// Q/K/V projections: C[m,n] = sum_d X[m,d] * W[n,d]   (y = x @ W^T)
// 2x2 register-tiled: one wave -> 32x32 output tile (4 WMMAs / K-step).
// K-loop chunked by 128: one uniform prefetch per 16 WMMAs, inner unrolled.
__global__ void proj_kernel(const float* __restrict__ X,
                            const float* __restrict__ Wq,
                            const float* __restrict__ Wk,
                            const float* __restrict__ Wv,
                            bf16* __restrict__ Qb,
                            bf16* __restrict__ Kb,
                            bf16* __restrict__ Vb) {
  const int lane = threadIdx.x & 31;
  const int wave = threadIdx.x >> 5;
  const int tile = blockIdx.x * 8 + wave;    // (4096/32)*(2048/32) = 8192 tiles
  const int mt = tile >> 6;                  // / (DM/32)
  const int nt = tile & 63;
  const float* W; bf16* O;
  if (blockIdx.y == 0)      { W = Wq; O = Qb; }
  else if (blockIdx.y == 1) { W = Wk; O = Kb; }
  else                      { W = Wv; O = Vb; }
  const int m0 = mt * 32, n0 = nt * 32;
  const int hi8  = (lane >> 4) << 3;
  const int hi16 = (lane >> 4) << 4;
  const float* arow0 = X + (size_t)(m0 + (lane & 15)) * DM;
  const float* arow1 = arow0 + (size_t)16 * DM;
  const float* brow0 = W + (size_t)(n0 + (lane & 15)) * DM;
  const float* brow1 = brow0 + (size_t)16 * DM;
  v8f acc00 = {}, acc01 = {}, acc10 = {}, acc11 = {};
  for (int kb = 0; kb < DM; kb += 128) {
    if (kb + 128 < DM) {                     // uniform, once per chunk
      __builtin_prefetch(arow0 + kb + 128, 0, 3);
      __builtin_prefetch(arow1 + kb + 128, 0, 3);
      __builtin_prefetch(brow0 + kb + 128, 0, 3);
      __builtin_prefetch(brow1 + kb + 128, 0, 3);
    }
#pragma unroll
    for (int k0 = kb; k0 < kb + 128; k0 += 32) {
      v16bf a0 = load_a_f32(arow0 + k0, hi8);
      v16bf a1 = load_a_f32(arow1 + k0, hi8);
      v16bf b0 = load_b_f32(brow0 + k0, hi16);
      v16bf b1 = load_b_f32(brow1 + k0, hi16);
      acc00 = wmma_bf16(a0, b0, acc00);
      acc01 = wmma_bf16(a0, b1, acc01);
      acc10 = wmma_bf16(a1, b0, acc10);
      acc11 = wmma_bf16(a1, b1, acc11);
    }
  }
  const int ncol  = n0 + (lane & 15);
  const int mbase = m0 + ((lane >> 4) << 3);
#pragma unroll
  for (int r = 0; r < 8; ++r) {
    O[(size_t)(mbase + r) * DM + ncol]           = (bf16)acc00[r];
    O[(size_t)(mbase + r) * DM + ncol + 16]      = (bf16)acc01[r];
    O[(size_t)(mbase + r + 16) * DM + ncol]      = (bf16)acc10[r];
    O[(size_t)(mbase + r + 16) * DM + ncol + 16] = (bf16)acc11[r];
  }
}

// ---------------------------------------------------------------- phase 2
// S[i,j] = (Q_i . K_j) / sqrt(DM); 2x2 tiled, lower-triangle 32x32 blocks.
__global__ void scores_kernel(const bf16* __restrict__ Qb,
                              const bf16* __restrict__ Kb,
                              float* __restrict__ S) {
  const int lane = threadIdx.x & 31;
  const int wave = threadIdx.x >> 5;
  const int rt2 = blockIdx.x;                 // 0..127 query 32-block
  const int jt2 = blockIdx.y * 8 + wave;      // 0..127 key 32-block
  if (jt2 > rt2) return;                      // causal block skip
  const int m0 = rt2 * 32, n0 = jt2 * 32;
  const int hi8  = (lane >> 4) << 3;
  const int hi16 = (lane >> 4) << 4;
  const bf16* arow0 = Qb + (size_t)(m0 + (lane & 15)) * DM;
  const bf16* arow1 = arow0 + (size_t)16 * DM;
  const bf16* brow0 = Kb + (size_t)(n0 + (lane & 15)) * DM;
  const bf16* brow1 = brow0 + (size_t)16 * DM;
  v8f acc00 = {}, acc01 = {}, acc10 = {}, acc11 = {};
  for (int kb = 0; kb < DM; kb += 128) {
    if (kb + 128 < DM) {
      __builtin_prefetch(arow0 + kb + 128, 0, 3);
      __builtin_prefetch(brow0 + kb + 128, 0, 3);
    }
#pragma unroll
    for (int k0 = kb; k0 < kb + 128; k0 += 32) {
      v16bf a0 = load_a_bf16(arow0 + k0, hi8);
      v16bf a1 = load_a_bf16(arow1 + k0, hi8);
      v16bf b0 = load_b_bf16(brow0 + k0, hi16);
      v16bf b1 = load_b_bf16(brow1 + k0, hi16);
      acc00 = wmma_bf16(a0, b0, acc00);
      acc01 = wmma_bf16(a0, b1, acc01);
      acc10 = wmma_bf16(a1, b0, acc10);
      acc11 = wmma_bf16(a1, b1, acc11);
    }
  }
  const float scale = 0.02209708691207961f;   // 1/sqrt(2048)
  const int ncol  = n0 + (lane & 15);
  const int mbase = m0 + ((lane >> 4) << 3);
#pragma unroll
  for (int r = 0; r < 8; ++r) {
    S[(size_t)(mbase + r) * SEQ + ncol]           = acc00[r] * scale;
    S[(size_t)(mbase + r) * SEQ + ncol + 16]      = acc01[r] * scale;
    S[(size_t)(mbase + r + 16) * SEQ + ncol]      = acc10[r] * scale;
    S[(size_t)(mbase + r + 16) * SEQ + ncol + 16] = acc11[r] * scale;
  }
}

// ---------------------------------------------------------------- phase 3
// Row softmax with exact causal mask; writes P in place (f32), covering
// columns [0, 16*ceil((r+1)/16)) -- exactly what phase 4 reads.
__global__ void softmax_kernel(float* __restrict__ S) {
  const int r = blockIdx.x;
  const int cEnd = ((r >> 4) + 1) << 4;
  float* row = S + (size_t)r * SEQ;
  __shared__ float sbuf[SEQ];
  __shared__ float red[256];
  const int t = threadIdx.x;

  float lmax = -1e30f;
  for (int c = t; c < cEnd; c += 256) {
    float v = (c <= r) ? row[c] : -1e30f;   // causal mask
    sbuf[c] = v;
    lmax = fmaxf(lmax, v);
  }
  red[t] = lmax; __syncthreads();
  for (int s = 128; s > 0; s >>= 1) {
    if (t < s) red[t] = fmaxf(red[t], red[t + s]);
    __syncthreads();
  }
  const float m = red[0]; __syncthreads();

  float lsum = 0.f;
  for (int c = t; c < cEnd; c += 256) {
    float e = __expf(sbuf[c] - m);
    sbuf[c] = e;
    lsum += e;
  }
  red[t] = lsum; __syncthreads();
  for (int s = 128; s > 0; s >>= 1) {
    if (t < s) red[t] += red[t + s];
    __syncthreads();
  }
  const float inv = 1.0f / red[0]; __syncthreads();

  for (int c = t; c < cEnd; c += 256) row[c] = sbuf[c] * inv;
}

// ---------------------------------------------------------------- phase 4
// O = P @ V. Block = 8 waves covering 16 rows x 256 cols; V tile (32x256)
// staged transposed in LDS; each wave does a 16x32 slab (2 WMMAs sharing
// one A fragment per K-step).
__global__ void av_kernel(const float* __restrict__ P,
                          const bf16* __restrict__ Vb,
                          float* __restrict__ Out) {
  const int lane = threadIdx.x & 31;
  const int wave = threadIdx.x >> 5;
  const int rt = blockIdx.x;           // query tile 0..255
  const int d0 = blockIdx.y * 256;     // output column chunk
  const int jCount = (rt + 1) * 16;    // causal key extent (tile aligned)
  const int m0 = rt * 16;
  const int hi8  = (lane >> 4) << 3;
  const int hi16 = (lane >> 4) << 4;
  const int nn0 = (wave << 5) + (lane & 15);   // wave's first 16-col group
  const int nn1 = nn0 + 16;

  __shared__ bf16 vt[256][40];         // [d_local][j_local], padded rows

  const float* prow = P + (size_t)(m0 + (lane & 15)) * SEQ;
  v8f acc0 = {}, acc1 = {};

  const int jl  = threadIdx.x >> 3;          // 0..31
  const int seg = (threadIdx.x & 7) * 32;    // 0..224

  int js = 0;
  for (; js + 32 <= jCount; js += 32) {
    // cooperative transposed load of V[js..js+31][d0..d0+255]
    const bf16* src = Vb + (size_t)(js + jl) * DM + d0 + seg;
    v8bf x0 = *(const v8bf*)src;
    v8bf x1 = *(const v8bf*)(src + 8);
    v8bf x2 = *(const v8bf*)(src + 16);
    v8bf x3 = *(const v8bf*)(src + 24);
#pragma unroll
    for (int i = 0; i < 8; ++i) {
      vt[seg + i][jl]      = x0[i];
      vt[seg + 8 + i][jl]  = x1[i];
      vt[seg + 16 + i][jl] = x2[i];
      vt[seg + 24 + i][jl] = x3[i];
    }
    __syncthreads();

    v16bf a = load_a_f32(prow + js, hi8);
    v8bf b00 = *(const v8bf*)&vt[nn0][hi16];
    v8bf b01 = *(const v8bf*)&vt[nn0][hi16 + 8];
    v8bf b10 = *(const v8bf*)&vt[nn1][hi16];
    v8bf b11 = *(const v8bf*)&vt[nn1][hi16 + 8];
    acc0 = wmma_bf16(a, cat16(b00, b01), acc0);
    acc1 = wmma_bf16(a, cat16(b10, b11), acc1);
    __syncthreads();
  }
  if (js < jCount) {                    // 16-column causal tail
    const bf16* src = Vb + (size_t)(js + jl) * DM + d0 + seg;
    v8bf x0 = *(const v8bf*)src;
    v8bf x1 = *(const v8bf*)(src + 8);
    v8bf x2 = *(const v8bf*)(src + 16);
    v8bf x3 = *(const v8bf*)(src + 24);
#pragma unroll
    for (int i = 0; i < 8; ++i) {
      vt[seg + i][jl]      = x0[i];
      vt[seg + 8 + i][jl]  = x1[i];
      vt[seg + 16 + i][jl] = x2[i];
      vt[seg + 24 + i][jl] = x3[i];
    }
    __syncthreads();

    v16bf a;
#pragma unroll
    for (int h = 0; h < 16; ++h)
      a[h] = (hi8 == 0) ? (bf16)prow[js + h] : (bf16)0.0f;
    v8bf b00 = *(const v8bf*)&vt[nn0][hi16];
    v8bf b01 = *(const v8bf*)&vt[nn0][hi16 + 8];
    v8bf b10 = *(const v8bf*)&vt[nn1][hi16];
    v8bf b11 = *(const v8bf*)&vt[nn1][hi16 + 8];
    acc0 = wmma_bf16(a, cat16(b00, b01), acc0);
    acc1 = wmma_bf16(a, cat16(b10, b11), acc1);
  }

  const int mbase = m0 + ((lane >> 4) << 3);
#pragma unroll
  for (int r = 0; r < 8; ++r) {
    Out[(size_t)(mbase + r) * DM + d0 + nn0] = acc0[r];
    Out[(size_t)(mbase + r) * DM + d0 + nn1] = acc1[r];
  }
}

// ---------------------------------------------------------------- launch
extern "C" void kernel_launch(void* const* d_in, const int* in_sizes, int n_in,
                              void* d_out, int out_size, void* d_ws, size_t ws_size,
                              hipStream_t stream) {
  (void)in_sizes; (void)n_in; (void)out_size; (void)ws_size;
  const float* X  = (const float*)d_in[0];
  const float* Wq = (const float*)d_in[1];
  const float* Wk = (const float*)d_in[2];
  const float* Wv = (const float*)d_in[3];
  // d_in[4] (mask) unused: causality is applied analytically.
  float* out = (float*)d_out;

  bf16* Qb = (bf16*)d_ws;                       // 16 MB
  bf16* Kb = Qb + (size_t)SEQ * DM;             // 16 MB
  bf16* Vb = Kb + (size_t)SEQ * DM;             // 16 MB
  float* S = (float*)(Vb + (size_t)SEQ * DM);   // 64 MB (S, then P in place)

  dim3 gProj((SEQ / 32) * (DM / 32) / 8, 3);    // 1024 x 3
  proj_kernel<<<gProj, 256, 0, stream>>>(X, Wq, Wk, Wv, Qb, Kb, Vb);

  dim3 gScore(SEQ / 32, SEQ / 32 / 8);          // 128 x 16
  scores_kernel<<<gScore, 256, 0, stream>>>(Qb, Kb, S);

  softmax_kernel<<<SEQ, 256, 0, stream>>>(S);

  dim3 gAV(SEQ / 16, DM / 256);                 // 256 x 8
  av_kernel<<<gAV, 256, 0, stream>>>(S, Vb, out);
}
